// MultiLayerPerceptronAttention_81621558493257
// MI455X (gfx1250) — matrix operand order repaired and checked by reading, not compiled, not run on hardware
//
#include <hip/hip_runtime.h>
#include <hip/hip_bf16.h>
#include <math.h>

// MI455X / gfx1250, wave32. Dominant op: (32*2048)x1024 @ 1024x1024 GEMM in
// bf16 WMMA; key/value stream 512MB from HBM (~22us floor at 23.3 TB/s).
// W1k repacked to 2MB fragment-native bf16 -> L2 resident.

typedef __attribute__((ext_vector_type(16))) __bf16 v16bf;
typedef __attribute__((ext_vector_type(8)))  float  v8f;

#define NEG_INF_F (-1000000000.0f)
#define B_  32
#define T_  2048
#define QD  1024
#define KD  1024
#define HD  1024
#define TW  1032   // 1024 + 8 bf16 pad: row stride 2064B = 516 dwords -> bank+4/row

#if __has_builtin(__builtin_amdgcn_tanhf)
__device__ __forceinline__ float fast_tanh(float x) { return __builtin_amdgcn_tanhf(x); }
#else
__device__ __forceinline__ float fast_tanh(float x) { return tanhf(x); }
#endif

union AFragU { v16bf v; uint4 q[2]; };
union BFragU { v16bf v; __bf16 e[16]; };

// ---------------------------------------------------------------- zero
__global__ void zero_kernel(float* scores, float* out) {
    int i = blockIdx.x * blockDim.x + threadIdx.x;
    if (i < B_ * T_) scores[i] = 0.0f;
    if (i < B_ * HD) out[i]    = 0.0f;
}

// ------------------------------------------------- pack W1k -> B fragments
// Fragment f = (ht, ks, lane): lane holds col n=lane&15, K = ks*32 + (lane>>4)*16 + [0..15]
// (contiguous K order shared with A, not the ISA table order: lanes 0-15 hold K=0..15, lanes 16-31 K=16..31,
//  VGPR j = K pair 2j,2j+1). 16 contiguous K per lane -> contiguous 16 floats in W1 row.
__global__ void pack_w1k_kernel(const float* __restrict__ W1, __bf16* __restrict__ w1kpk) {
    int gid  = blockIdx.x * blockDim.x + threadIdx.x;   // 65536 fragment-lanes
    int lane = gid & 31;
    int ks   = (gid >> 5) & 31;
    int ht   = gid >> 10;                               // 0..63
    int n    = lane & 15;
    int kb   = (lane >> 4) * 16;
    int h    = ht * 16 + n;
    const float* src = W1 + (size_t)h * (QD + KD) + QD + ks * 32 + kb;
    BFragU f;
#pragma unroll
    for (int i = 0; i < 16; ++i) f.e[i] = (__bf16)src[i];
    *reinterpret_cast<v16bf*>(w1kpk + (size_t)gid * 16) = f.v;
}

// ---------------------------------------------------------------- qproj
__global__ void qproj_kernel(const float* __restrict__ query,
                             const float* __restrict__ W1,
                             float* __restrict__ qproj) {
    int gid = blockIdx.x * blockDim.x + threadIdx.x;    // 32768
    int b = gid >> 10, h = gid & 1023;
    const float* q = query + (size_t)b * QD;
    const float* w = W1 + (size_t)h * (QD + KD);
    float s = 0.0f;
    for (int i = 0; i < QD; ++i) s = fmaf(q[i], w[i], s);
    qproj[gid] = s;
}

// --------------------------------------------- WMMA kproj+tanh+score kernel
// Block: 8 waves, 32 t-rows of one batch, full H=1024. Wave w -> h-tiles [8w,8w+8).
__global__ __launch_bounds__(256, 1)
void score_kernel(const float* __restrict__ key,
                  const float* __restrict__ qproj,
                  const float* __restrict__ W2,
                  const __bf16* __restrict__ w1kpk,
                  float* __restrict__ scores) {
    __shared__ __bf16 sA[32 * TW];
    const int b   = blockIdx.y;
    const int t0  = blockIdx.x * 32;
    const int tid = threadIdx.x;

    // Stage key tile (32 x 1024) as bf16 into LDS, coalesced f32 reads.
    const float* krow = key + ((size_t)b * T_ + t0) * KD;
    for (int idx = tid; idx < 32 * KD; idx += 256) {
        int r = idx >> 10, c = idx & 1023;
        sA[r * TW + c] = (__bf16)krow[(size_t)r * KD + c];
    }
    __syncthreads();

    const int w    = tid >> 5;        // wave id 0..7
    const int lane = tid & 31;
    const int n    = lane & 15;       // A row within tile / C column / B column
    const int kg   = lane >> 4;       // lane half-group
    const int k0   = kg * 16;         // A: lanes>=16 hold K=16..31 contiguous

    // Single per-wave B base pointer; j/ks terms become constant instruction offsets.
    // frag(ht=w*8+j, ks, lane) element offset = (w*8192 + lane)*16 + j*16384 + ks*512
    const __bf16* bptr = w1kpk + ((size_t)w * 8192 + lane) * 16;

    v8f acc[2][8] = {};               // 2 t-tiles x 8 h-tiles, 16x16 f32 each

    for (int ks = 0; ks < 32; ++ks) {
        AFragU a0, a1;
        int base = n * TW + ks * 32 + k0;
        a0.q[0] = *reinterpret_cast<const uint4*>(&sA[base]);
        a0.q[1] = *reinterpret_cast<const uint4*>(&sA[base + 8]);
        a1.q[0] = *reinterpret_cast<const uint4*>(&sA[base + 16 * TW]);
        a1.q[1] = *reinterpret_cast<const uint4*>(&sA[base + 16 * TW + 8]);
        const __bf16* bks = bptr + ks * 512;
#pragma unroll
        for (int j = 0; j < 8; ++j) {
            v16bf bf = *reinterpret_cast<const v16bf*>(bks + j * 16384);
            acc[0][j] = __builtin_amdgcn_wmma_f32_16x16x32_bf16(
                false, a0.v, false, bf, (short)0, acc[0][j], false, false);
            acc[1][j] = __builtin_amdgcn_wmma_f32_16x16x32_bf16(
                false, a1.v, false, bf, (short)0, acc[1][j], false, false);
        }
    }

    // Epilogue: C layout -> VGPR r holds row M = r + 8*kg, column N = n.
    // score[b,t] partial = sum_h tanh(kproj + qproj[b,h]) * W2[h]
#pragma unroll
    for (int tt = 0; tt < 2; ++tt) {
        float srow[8];
#pragma unroll
        for (int r = 0; r < 8; ++r) srow[r] = 0.0f;
#pragma unroll
        for (int j = 0; j < 8; ++j) {
            int h    = w * 128 + j * 16 + n;
            float qb = qproj[(size_t)b * HD + h];
            float w2 = W2[h];
#pragma unroll
            for (int r = 0; r < 8; ++r)
                srow[r] += fast_tanh(acc[tt][j][r] + qb) * w2;
        }
#pragma unroll
        for (int r = 0; r < 8; ++r) {
            float v = srow[r];
            v += __shfl_xor(v, 1);
            v += __shfl_xor(v, 2);
            v += __shfl_xor(v, 4);
            v += __shfl_xor(v, 8);    // reduce 16-lane group (bit4 untouched)
            if (n == 0)
                atomicAdd(&scores[(size_t)b * T_ + t0 + tt * 16 + kg * 8 + r], v);
        }
    }
}

// ---------------------------------------------------------------- softmax
__global__ void softmax_kernel(const float* __restrict__ scores,
                               const int* __restrict__ mask,
                               float* __restrict__ prob) {
    int b = blockIdx.x, tid = threadIdx.x;
    __shared__ float red[256];
    float mx = NEG_INF_F;
    for (int t = tid; t < T_; t += 256) {
        float s = (mask[b * T_ + t] == 0) ? NEG_INF_F : scores[b * T_ + t];
        mx = fmaxf(mx, s);
    }
    red[tid] = mx; __syncthreads();
    for (int s = 128; s > 0; s >>= 1) {
        if (tid < s) red[tid] = fmaxf(red[tid], red[tid + s]);
        __syncthreads();
    }
    mx = red[0]; __syncthreads();
    float sum = 0.0f;
    for (int t = tid; t < T_; t += 256) {
        float s = (mask[b * T_ + t] == 0) ? NEG_INF_F : scores[b * T_ + t];
        float e = __expf(s - mx);
        prob[b * T_ + t] = e;
        sum += e;
    }
    red[tid] = sum; __syncthreads();
    for (int s = 128; s > 0; s >>= 1) {
        if (tid < s) red[tid] += red[tid + s];
        __syncthreads();
    }
    float inv = 1.0f / red[0];
    for (int t = tid; t < T_; t += 256) prob[b * T_ + t] *= inv;
}

// ------------------------------------------------------- out = prob @ value
__global__ void out_kernel(const float* __restrict__ prob,
                           const float* __restrict__ value,
                           float* __restrict__ out) {
    int b = blockIdx.x, chunk = blockIdx.y, tid = threadIdx.x;
    __shared__ float sp[256];
    int t0 = chunk * 256;
    sp[tid] = prob[(size_t)b * T_ + t0 + tid];
    __syncthreads();
    int v = tid * 4;
    float4 acc = make_float4(0.f, 0.f, 0.f, 0.f);
    for (int i = 0; i < 256; ++i) {
        float p = sp[i];
        const float4 val = *reinterpret_cast<const float4*>(
            value + ((size_t)b * T_ + t0 + i) * KD + v);
        acc.x = fmaf(p, val.x, acc.x);
        acc.y = fmaf(p, val.y, acc.y);
        acc.z = fmaf(p, val.z, acc.z);
        acc.w = fmaf(p, val.w, acc.w);
    }
    atomicAdd(&out[(size_t)b * HD + v + 0], acc.x);
    atomicAdd(&out[(size_t)b * HD + v + 1], acc.y);
    atomicAdd(&out[(size_t)b * HD + v + 2], acc.z);
    atomicAdd(&out[(size_t)b * HD + v + 3], acc.w);
}

// ---------------------------------------------------------------- launch
extern "C" void kernel_launch(void* const* d_in, const int* in_sizes, int n_in,
                              void* d_out, int out_size, void* d_ws, size_t ws_size,
                              hipStream_t stream) {
    (void)in_sizes; (void)n_in; (void)out_size; (void)ws_size;
    const float* query = (const float*)d_in[0];
    const float* key   = (const float*)d_in[1];
    const float* value = (const float*)d_in[2];
    const int*   mask  = (const int*)d_in[3];
    const float* W1    = (const float*)d_in[4];
    const float* W2    = (const float*)d_in[5];
    float* out = (float*)d_out;

    // workspace layout (~2.65 MB total)
    float*  qproj  = (float*)d_ws;            // 32*1024 f32
    float*  scores = qproj + B_ * HD;         // 32*2048 f32
    float*  prob   = scores + B_ * T_;        // 32*2048 f32
    __bf16* w1kpk  = (__bf16*)(prob + B_ * T_); // 64*32*32*16 bf16 = 2MB

    zero_kernel   <<<256, 256, 0, stream>>>(scores, out);
    pack_w1k_kernel<<<256, 256, 0, stream>>>(W1, w1kpk);
    qproj_kernel  <<<128, 256, 0, stream>>>(query, W1, qproj);
    score_kernel  <<<dim3(T_ / 32, B_), 256, 0, stream>>>(key, qproj, W2, w1kpk, scores);
    softmax_kernel<<<B_, 256, 0, stream>>>(scores, mask, prob);
    out_kernel    <<<dim3(B_, T_ / 256), 256, 0, stream>>>(prob, value, out);
}
